// Alignn_59768764891855
// MI455X (gfx1250) — matrix operand order repaired and checked by reading, not compiled, not run on hardware
//
#include <hip/hip_runtime.h>
#include <hip/hip_bf16.h>

// ---------------------------------------------------------------------------
// SchNet-style GNN for MI455X (gfx1250), wave32 + WMMA bf16.
// Fused per-layer edge kernel: rbf -> GEMM(Wf1) -> ssp -> GEMM(Wf2) -> gate
// -> atomic scatter, with [E,128] intermediates never touching HBM.
// ---------------------------------------------------------------------------

typedef __attribute__((ext_vector_type(16))) __bf16 v16bf;
typedef __attribute__((ext_vector_type(2)))  __bf16 v2bf;
typedef __attribute__((ext_vector_type(8)))  float  v8f;

static constexpr int N_ = 50000, E_ = 800000, F_ = 64, H_ = 128, G_ = 64, C_ = 10;

// --- bf16 weight-fragment arena offsets (in bf16 elements) ---
// Each 32x16 B-tile is 512 bf16, stored per-lane contiguous (lane*16 .. +15).
static constexpr long FR_W1_0 = 0;                   // K=64  -> 16 tiles
static constexpr long FR_W1_R = 8192;                // 3 mats, 32 tiles each
static constexpr long FR_WF1  = FR_W1_R + 3 * 16384; // 57344
static constexpr long FR_WF2  = FR_WF1 + 4 * 16384;  // 122880
static constexpr long FR_W2   = FR_WF2 + 4 * 16384;  // 188416

// --- workspace byte offsets ---
static constexpr long WS_FRAGS = 0;          // 507904 B (pad to 512K)
static constexpr long WS_XB    = 524288;     // N*64 bf16  = 6.4 MB
static constexpr long WS_HIN   = 6924288;    // N*128 bf16 = 12.8 MB
static constexpr long WS_HPROJ = 19724288;   // N*128 bf16 = 12.8 MB
static constexpr long WS_AGG   = 32524288;   // N*128 f32  = 25.6 MB
static constexpr long WS_HFIN  = 58124288;   // N*128 f32  = 25.6 MB
static constexpr long WS_POOL  = 83724288;   // G*128 f32

// ---------------------------------------------------------------------------
// WMMA helpers
// ---------------------------------------------------------------------------
__device__ inline v8f wmma_bf16(v16bf a, v16bf b, v8f c) {
  return __builtin_amdgcn_wmma_f32_16x16x32_bf16(false, a, false, b,
                                                 (short)0, c, false, false);
}

// A-matrix 16x32 bf16 fragment from a row-major source (K-pairs contiguous).
// ISA layout: lane m=L&15, half=L>>4; VGPR p<4: K=2p+8*half(+0/1);
//             p>=4: K=16+2(p-4)+8*half(+0/1).
__device__ inline v16bf load_a_bf(const __bf16* __restrict__ base, int ld, int lane) {
  int m = lane & 15, half = lane >> 4;
  const __bf16* row = base + (long)m * ld;
  v16bf a;
#pragma unroll
  for (int p = 0; p < 8; ++p) {
    int k = ((p < 4) ? 2 * p : 16 + 2 * (p - 4)) + half * 8;
    v2bf pr = *(const v2bf*)(row + k);
    a[2 * p] = pr.x;
    a[2 * p + 1] = pr.y;
  }
  return a;
}

// Same, but converting f32 source on the fly.
__device__ inline v16bf load_a_f32(const float* __restrict__ base, int ld, int lane) {
  int m = lane & 15, half = lane >> 4;
  const float* row = base + (long)m * ld;
  v16bf a;
#pragma unroll
  for (int p = 0; p < 8; ++p) {
    int k = ((p < 4) ? 2 * p : 16 + 2 * (p - 4)) + half * 8;
    float2 pr = *(const float2*)(row + k);
    a[2 * p] = (__bf16)pr.x;
    a[2 * p + 1] = (__bf16)pr.y;
  }
  return a;
}

// Pre-swizzled B fragment: per-lane contiguous 16 bf16 (two b128 loads).
__device__ inline v16bf load_b(const __bf16* __restrict__ fragbase, int tile, int lane) {
  return *(const v16bf*)(fragbase + (long)tile * 512 + lane * 16);
}

__device__ inline float ssp(float v) {
  float sp = (v > 15.0f) ? v : __logf(1.0f + __expf(v));
  return sp - 0.6931472f;
}

// ---------------------------------------------------------------------------
// Weight prep: f32 -> bf16 B-fragment arena, plus x -> bf16
// ---------------------------------------------------------------------------
__device__ inline void swz_mat(__bf16* dst, const float* src, int KT,
                               long tid, long stride) {
  long total = (long)KT * 8 * 512;
  for (long i = tid; i < total; i += stride) {
    int t  = (int)(i >> 9);
    int r  = (int)(i & 511);
    int Ln = r >> 4;
    int e  = r & 15;
    int kt = t >> 3;
    int nt = t & 7;
    int k  = 32 * kt + ((e < 8) ? 8 * (Ln >> 4) + e : 16 + 8 * (Ln >> 4) + (e - 8));   // B 32x16 bf16: K = 8*half + e (e<8), 16 + 8*half + (e-8) (e>=8)
    int n  = 16 * nt + (Ln & 15);
    dst[i] = (__bf16)src[(long)k * H_ + n];
  }
}

__global__ void prep_kernel(const float* __restrict__ x,
                            const float* __restrict__ W1_0,
                            const float* __restrict__ W1_rest,
                            const float* __restrict__ Wf1,
                            const float* __restrict__ Wf2,
                            const float* __restrict__ W2,
                            __bf16* __restrict__ frags,
                            __bf16* __restrict__ xb) {
  long tid = (long)blockIdx.x * blockDim.x + threadIdx.x;
  long stride = (long)gridDim.x * blockDim.x;
  for (long i = tid; i < (long)N_ * F_; i += stride) xb[i] = (__bf16)x[i];
  swz_mat(frags + FR_W1_0, W1_0, 2, tid, stride);
  for (int i = 0; i < 3; ++i)
    swz_mat(frags + FR_W1_R + i * 16384, W1_rest + (long)i * H_ * H_, 4, tid, stride);
  for (int l = 0; l < 4; ++l) {
    swz_mat(frags + FR_WF1 + l * 16384, Wf1 + (long)l * H_ * H_, 4, tid, stride);
    swz_mat(frags + FR_WF2 + l * 16384, Wf2 + (long)l * H_ * H_, 4, tid, stride);
    swz_mat(frags + FR_W2  + l * 16384, W2  + (long)l * H_ * H_, 4, tid, stride);
  }
}

__global__ void zero_f32(float* __restrict__ p, long n) {
  long i = (long)blockIdx.x * blockDim.x + threadIdx.x;
  if (i < n) p[i] = 0.0f;
}

// ---------------------------------------------------------------------------
// Node projection: out_bf[N,128] = bf16(in[N,K] @ W1 + b1)
// One wave per 16-row tile; 8 col-tiles x KT k-tiles of WMMA.
// ---------------------------------------------------------------------------
template <int K>
__global__ __launch_bounds__(256) void node_proj_kernel(
    const __bf16* __restrict__ in, const __bf16* __restrict__ wfrag,
    const float* __restrict__ bias, __bf16* __restrict__ out) {
  int wave = (int)(((long)blockIdx.x * blockDim.x + threadIdx.x) >> 5);
  int lane = threadIdx.x & 31;
  if (wave >= N_ / 16) return;
  constexpr int KT = K / 32;
  const __bf16* abase = in + (long)wave * 16 * K;
  v16bf a[KT];
#pragma unroll
  for (int kt = 0; kt < KT; ++kt) a[kt] = load_a_bf(abase + 32 * kt, K, lane);
  int m = lane & 15, half = lane >> 4;
#pragma unroll
  for (int nt = 0; nt < 8; ++nt) {
    v8f acc = {};
#pragma unroll
    for (int kt = 0; kt < KT; ++kt)
      acc = wmma_bf16(a[kt], load_b(wfrag, kt * 8 + nt, lane), acc);
    float bv = bias[16 * nt + m];
#pragma unroll
    for (int r = 0; r < 8; ++r) {
      long row = (long)wave * 16 + r + 8 * half;
      out[row * H_ + 16 * nt + m] = (__bf16)(acc[r] + bv);
    }
  }
}

// ---------------------------------------------------------------------------
// Output projection: h = relu(agg[N,128] @ W2 + b2) -> bf16 (+f32 on last
// layer, selected at compile time to keep the epilogue branch-free).
// ---------------------------------------------------------------------------
template <bool STORE_F32>
__global__ __launch_bounds__(256) void out_proj_kernel(
    const float* __restrict__ agg, const __bf16* __restrict__ wfrag,
    const float* __restrict__ bias, __bf16* __restrict__ out_bf,
    float* __restrict__ out_f32) {
  int wave = (int)(((long)blockIdx.x * blockDim.x + threadIdx.x) >> 5);
  int lane = threadIdx.x & 31;
  if (wave >= N_ / 16) return;
  const float* abase = agg + (long)wave * 16 * H_;
  v16bf a[4];
#pragma unroll
  for (int kt = 0; kt < 4; ++kt) a[kt] = load_a_f32(abase + 32 * kt, H_, lane);
  int m = lane & 15, half = lane >> 4;
#pragma unroll
  for (int nt = 0; nt < 8; ++nt) {
    v8f acc = {};
#pragma unroll
    for (int kt = 0; kt < 4; ++kt)
      acc = wmma_bf16(a[kt], load_b(wfrag, kt * 8 + nt, lane), acc);
    float bv = bias[16 * nt + m];
#pragma unroll
    for (int r = 0; r < 8; ++r) {
      float v = acc[r] + bv;
      v = fmaxf(v, 0.0f);
      long idx = ((long)wave * 16 + r + 8 * half) * H_ + 16 * nt + m;
      out_bf[idx] = (__bf16)v;
      if constexpr (STORE_F32) out_f32[idx] = v;
    }
  }
}

// ---------------------------------------------------------------------------
// Fused CFConv edge kernel: one wave per 16-edge tile.
//   rbf(16x128, computed in-flight) -> GEMM(Wf1) -> ssp -> GEMM(Wf2)
//   -> gate with hproj[src] -> atomic scatter-add into agg[dst].
// LDS used to relayout C-fragments -> A-fragments between GEMMs.
// ---------------------------------------------------------------------------
__global__ __launch_bounds__(256) void edge_kernel(
    const float* __restrict__ dist, const int* __restrict__ esrc,
    const int* __restrict__ edst, const __bf16* __restrict__ hproj,
    const __bf16* __restrict__ wf1, const __bf16* __restrict__ wf2,
    const float* __restrict__ bf1, const float* __restrict__ bf2,
    float* __restrict__ agg) {
  __shared__ __align__(16) __bf16 stage[8][16 * H_];  // 4 KB per wave
  int wid = threadIdx.x >> 5;
  int lane = threadIdx.x & 31;
  long tile = (long)blockIdx.x * 8 + wid;
  if (tile >= E_ / 16) return;
  long e0 = tile * 16;
  __bf16* st = stage[wid];
  int m = lane & 15, half = lane >> 4;

  // Edge endpoints for the 8 rows this lane will gate/scatter (M = r + 8*half).
  int es[8], ed[8];
#pragma unroll
  for (int r = 0; r < 8; ++r) {
    int e = r + 8 * half;
    es[r] = esrc[e0 + e];
    ed[r] = edst[e0 + e];
  }

  // Gaussian RBF expansion into LDS: lane covers edge m, 64 centers.
  float d = dist[e0 + m];
#pragma unroll
  for (int j0 = 0; j0 < 64; ++j0) {
    int j = 64 * half + j0;
    float v = d - (float)j * (1.0f / 127.0f);
    st[m * H_ + j] = (__bf16)__expf(-10.0f * v * v);
  }
  asm volatile("s_wait_dscnt 0x0" ::: "memory");

  v16bf a[4];
#pragma unroll
  for (int kt = 0; kt < 4; ++kt) a[kt] = load_a_bf(st + 32 * kt, H_, lane);

  v8f acc[8];
#pragma unroll
  for (int nt = 0; nt < 8; ++nt) {
    v8f c = {};
#pragma unroll
    for (int kt = 0; kt < 4; ++kt)
      c = wmma_bf16(a[kt], load_b(wf1, kt * 8 + nt, lane), c);
    acc[nt] = c;
  }

  // ssp + bias, relayout C -> LDS (M = r+8*half, N = 16*nt + m).
  asm volatile("s_wait_dscnt 0x0" ::: "memory");
#pragma unroll
  for (int nt = 0; nt < 8; ++nt) {
    float bv = bf1[16 * nt + m];
#pragma unroll
    for (int r = 0; r < 8; ++r) {
      float v = ssp(acc[nt][r] + bv);
      st[(r + 8 * half) * H_ + 16 * nt + m] = (__bf16)v;
    }
  }
  asm volatile("s_wait_dscnt 0x0" ::: "memory");

#pragma unroll
  for (int kt = 0; kt < 4; ++kt) a[kt] = load_a_bf(st + 32 * kt, H_, lane);
#pragma unroll
  for (int nt = 0; nt < 8; ++nt) {
    v8f c = {};
#pragma unroll
    for (int kt = 0; kt < 4; ++kt)
      c = wmma_bf16(a[kt], load_b(wf2, kt * 8 + nt, lane), c);
    acc[nt] = c;
  }

  // Gate with gathered source features; scatter-add to destinations.
#pragma unroll
  for (int nt = 0; nt < 8; ++nt) {
    int n = 16 * nt + m;
    float bv = bf2[n];
#pragma unroll
    for (int r = 0; r < 8; ++r) {
      float filt = acc[nt][r] + bv;
      float hv = (float)hproj[(long)es[r] * H_ + n];
      atomicAdd(&agg[(long)ed[r] * H_ + n], hv * filt);
    }
  }
}

// ---------------------------------------------------------------------------
// Per-graph mean pooling (graph_ids sorted -> binary search, no atomics)
// ---------------------------------------------------------------------------
__device__ inline int lower_bound_i32(const int* a, int n, int v) {
  int lo = 0, hi = n;
  while (lo < hi) {
    int mid = (lo + hi) >> 1;
    if (a[mid] < v) lo = mid + 1; else hi = mid;
  }
  return lo;
}

__global__ void pool_kernel(const float* __restrict__ hfin,
                            const int* __restrict__ gids,
                            float* __restrict__ pooled) {
  int g = blockIdx.x;
  int j = threadIdx.x;  // 128 threads = one column each
  int s = lower_bound_i32(gids, N_, g);
  int e = lower_bound_i32(gids, N_, g + 1);
  float sum = 0.0f;
  for (int i = s; i < e; ++i) sum += hfin[(long)i * H_ + j];
  float cnt = fmaxf((float)(e - s), 1.0f);
  pooled[g * H_ + j] = sum / cnt;
}

__global__ void head_kernel(const float* __restrict__ pooled,
                            const float* __restrict__ fcw,
                            const float* __restrict__ fcb,
                            float* __restrict__ out) {
  int g = threadIdx.x;
  if (g >= G_) return;
  float logits[C_];
#pragma unroll
  for (int c = 0; c < C_; ++c) logits[c] = fcb[c];
  for (int h = 0; h < H_; ++h) {
    float p = pooled[g * H_ + h];
#pragma unroll
    for (int c = 0; c < C_; ++c) logits[c] += p * fcw[h * C_ + c];
  }
  float mx = logits[0];
#pragma unroll
  for (int c = 1; c < C_; ++c) mx = fmaxf(mx, logits[c]);
  float sum = 0.0f;
#pragma unroll
  for (int c = 0; c < C_; ++c) sum += __expf(logits[c] - mx);
  float lse = mx + __logf(sum);
#pragma unroll
  for (int c = 0; c < C_; ++c) out[g * C_ + c] = logits[c] - lse;
}

// ---------------------------------------------------------------------------
// Launch
// ---------------------------------------------------------------------------
extern "C" void kernel_launch(void* const* d_in, const int* in_sizes, int n_in,
                              void* d_out, int out_size, void* d_ws, size_t ws_size,
                              hipStream_t stream) {
  (void)in_sizes; (void)n_in; (void)out_size; (void)ws_size;
  const float* x       = (const float*)d_in[0];
  const float* edist   = (const float*)d_in[1];
  const int*   esrc    = (const int*)d_in[2];
  const int*   edst    = (const int*)d_in[3];
  const int*   gids    = (const int*)d_in[4];
  const float* W1_0    = (const float*)d_in[5];
  const float* b1_0    = (const float*)d_in[6];
  const float* W1_rest = (const float*)d_in[7];
  const float* b1_rest = (const float*)d_in[8];
  const float* Wf1     = (const float*)d_in[9];
  const float* bf1     = (const float*)d_in[10];
  const float* Wf2     = (const float*)d_in[11];
  const float* bf2     = (const float*)d_in[12];
  const float* W2      = (const float*)d_in[13];
  const float* b2      = (const float*)d_in[14];
  const float* fcw     = (const float*)d_in[15];
  const float* fcb     = (const float*)d_in[16];

  char* ws = (char*)d_ws;
  __bf16* frags  = (__bf16*)(ws + WS_FRAGS);
  __bf16* xb     = (__bf16*)(ws + WS_XB);
  __bf16* hin    = (__bf16*)(ws + WS_HIN);
  __bf16* hproj  = (__bf16*)(ws + WS_HPROJ);
  float*  agg    = (float*)(ws + WS_AGG);
  float*  hfin   = (float*)(ws + WS_HFIN);
  float*  pooled = (float*)(ws + WS_POOL);

  prep_kernel<<<1024, 256, 0, stream>>>(x, W1_0, W1_rest, Wf1, Wf2, W2, frags, xb);

  const int node_blocks = (N_ / 16 + 7) / 8;   // 391 (8 waves/block)
  const int edge_blocks = (E_ / 16) / 8;       // 6250
  const long aggn = (long)N_ * H_;

  for (int l = 0; l < 4; ++l) {
    if (l == 0)
      node_proj_kernel<64><<<node_blocks, 256, 0, stream>>>(
          xb, frags + FR_W1_0, b1_0, hproj);
    else
      node_proj_kernel<128><<<node_blocks, 256, 0, stream>>>(
          hin, frags + FR_W1_R + (l - 1) * 16384, b1_rest + (long)(l - 1) * H_, hproj);

    zero_f32<<<(int)((aggn + 255) / 256), 256, 0, stream>>>(agg, aggn);

    edge_kernel<<<edge_blocks, 256, 0, stream>>>(
        edist, esrc, edst, hproj,
        frags + FR_WF1 + l * 16384, frags + FR_WF2 + l * 16384,
        bf1 + (long)l * H_, bf2 + (long)l * H_, agg);

    if (l == 3)
      out_proj_kernel<true><<<node_blocks, 256, 0, stream>>>(
          agg, frags + FR_W2 + l * 16384, b2 + (long)l * H_, hin, hfin);
    else
      out_proj_kernel<false><<<node_blocks, 256, 0, stream>>>(
          agg, frags + FR_W2 + l * 16384, b2 + (long)l * H_, hin, hfin);
  }

  pool_kernel<<<G_, H_, 0, stream>>>(hfin, gids, pooled);
  head_kernel<<<1, 64, 0, stream>>>(pooled, fcw, fcb, (float*)d_out);
}
